// RNNModel_20014547599695
// MI455X (gfx1250) — compile-verified, matrix operand order checked by reading, not run on hardware
//
#include <hip/hip_runtime.h>

// Fused RNN forward for MI455X (gfx1250, wave32, WMMA).
// One wave owns a 16-row batch tile across all 256 timesteps.
// Per step: 12x v_wmma_f32_16x16x32_f16 (f16 in, f32 acc), v_tanh_f32,
// C/D->A relayout through wave-private padded LDS.
// Software-pipelined: x(t+1) load + its input-projection WMMAs are issued
// during step t (off the recurrence critical path); the two K-halves of
// h@W_hh^T use independent accumulator banks to avoid serial WMMA chains.

typedef __attribute__((ext_vector_type(16))) _Float16 v16h;
typedef __attribute__((ext_vector_type(8)))  float    v8f;
typedef __attribute__((ext_vector_type(4)))  float    v4f;

#define RNN_BATCH       4096
#define RNN_SEQ         256
#define RNN_INPUT       32
#define RNN_HIDDEN      64
#define ROWS_PER_WAVE   16
#define WAVES_PER_BLOCK 4
#define LDS_STRIDE      68   // 64 + 4 floats pad; 68*4B = 272B keeps 16B alignment

__device__ __forceinline__ float tanh_fast(float x) {
#if __has_builtin(__builtin_amdgcn_tanhf)
  return __builtin_amdgcn_tanhf(x);
#elif __has_builtin(__builtin_amdgcn_tanh_f32)
  return __builtin_amdgcn_tanh_f32(x);
#else
  float e = __builtin_amdgcn_exp2f(x * 2.8853900817779268f);
  return 1.0f - 2.0f / (1.0f + e);
#endif
}

// Load 16 consecutive f32 (16B aligned) and convert to 16 packed f16.
__device__ __forceinline__ v16h load16_cvt_f16(const float* p) {
  v16h r;
#pragma unroll
  for (int i = 0; i < 4; ++i) {
    v4f f = *(const v4f*)(p + 4 * i);
#pragma unroll
    for (int j = 0; j < 4; ++j) r[4 * i + j] = (_Float16)f[j];
  }
  return r;
}

__global__ __launch_bounds__(WAVES_PER_BLOCK * 32)
void rnn_fused_kernel(const float* __restrict__ x,
                      const float* __restrict__ W_ih,
                      const float* __restrict__ W_hh,
                      const float* __restrict__ b_ih,
                      const float* __restrict__ b_hh,
                      const float* __restrict__ W_fc,
                      const float* __restrict__ b_fc,
                      float* __restrict__ out) {
  __shared__ float hbuf_all[WAVES_PER_BLOCK][ROWS_PER_WAVE * LDS_STRIDE];

  const int tid  = threadIdx.x;
  const int wave = tid >> 5;
  const int lane = tid & 31;
  const int m    = lane & 15;  // M row (A) / N column (B) within 16-tile
  const int kg   = lane >> 4;  // K half-group (0/1) per WMMA f16 layout

  float* hbuf = hbuf_all[wave];

  const int tile = blockIdx.x * WAVES_PER_BLOCK + wave;
  const int row0 = tile * ROWS_PER_WAVE;

  // ---- stationary B fragments: W_ih^T tiles [K=32, N=16] x 4 ----
  // B layout: lane L holds column N = L%16, K = (L/16)*16 + j.  B[k][n] = W[n][k].
  v16h wihB[4];
#pragma unroll
  for (int nt = 0; nt < 4; ++nt)
    wihB[nt] = load16_cvt_f16(W_ih + (size_t)(nt * 16 + m) * RNN_INPUT + kg * 16);

  // W_hh^T tiles: kc in {0,1} (K chunk of 32), nt in {0..3} (N tile of 16)
  v16h whhB[2][4];
#pragma unroll
  for (int kc = 0; kc < 2; ++kc)
#pragma unroll
    for (int nt = 0; nt < 4; ++nt)
      whhB[kc][nt] =
          load16_cvt_f16(W_hh + (size_t)(nt * 16 + m) * RNN_HIDDEN + kc * 32 + kg * 16);

  // Constant C operands: bias (b_ih + b_hh) splat per N-tile, and zero.
  // In C/D layout the hidden index n = nt*16 + lane%16 is VGPR-invariant,
  // so the bias accumulator seed is a per-lane splat, built once.
  v8f biasC[4];
#pragma unroll
  for (int nt = 0; nt < 4; ++nt) {
    float b = b_ih[nt * 16 + m] + b_hh[nt * 16 + m];
#pragma unroll
    for (int r = 0; r < 8; ++r) biasC[nt][r] = b;
  }
  v8f zeroC;
#pragma unroll
  for (int r = 0; r < 8; ++r) zeroC[r] = 0.0f;

  // ---- recurrent state h as f16 A fragments, h0 = 0 ----
  v16h hA[2];
#pragma unroll
  for (int kc = 0; kc < 2; ++kc)
#pragma unroll
    for (int j = 0; j < 16; ++j) hA[kc][j] = (_Float16)0.0f;

  // A-fragment source for x: lane holds row (row0 + m), K = kg*16 + j
  const float* xlane = x + ((size_t)(row0 + m) * RNN_SEQ) * RNN_INPUT + kg * 16;

  // ---- software pipeline prologue: x(0) projection ----
  v16h xA = load16_cvt_f16(xlane);
  v8f xpart[4];
#pragma unroll
  for (int nt = 0; nt < 4; ++nt)
    xpart[nt] = __builtin_amdgcn_wmma_f32_16x16x32_f16(
        false, xA, false, wihB[nt], (short)0, zeroC, false, false);

  for (int t = 0; t < RNN_SEQ; ++t) {
    // Issue next x load early (clamped to avoid OOB on the last step).
    const int tn = (t + 1 < RNN_SEQ) ? (t + 1) : t;
    v16h xA_next = load16_cvt_f16(xlane + tn * RNN_INPUT);

    // Recurrent GEMM: two independent accumulator banks (kc=0 / kc=1)
    // so the per-step WMMA chain depth is 1, not 3.
    v8f accA[4], accB[4];
#pragma unroll
    for (int nt = 0; nt < 4; ++nt)
      accA[nt] = __builtin_amdgcn_wmma_f32_16x16x32_f16(
          false, hA[0], false, whhB[0][nt], (short)0, biasC[nt], false, false);
#pragma unroll
    for (int nt = 0; nt < 4; ++nt)
      accB[nt] = __builtin_amdgcn_wmma_f32_16x16x32_f16(
          false, hA[1], false, whhB[1][nt], (short)0, xpart[nt], false, false);

    // Next step's input projection: h-independent, fills hazard slots.
    v8f xpart_next[4];
#pragma unroll
    for (int nt = 0; nt < 4; ++nt)
      xpart_next[nt] = __builtin_amdgcn_wmma_f32_16x16x32_f16(
          false, xA_next, false, wihB[nt], (short)0, zeroC, false, false);

    // h = tanh(accA + accB): C/D layout -> row-major h in wave-private LDS.
    // C/D: lane L, VGPR r holds element [m_row = r + 8*(L/16)][n = nt*16 + L%16]
#pragma unroll
    for (int nt = 0; nt < 4; ++nt)
#pragma unroll
      for (int r = 0; r < 8; ++r)
        hbuf[(r + 8 * kg) * LDS_STRIDE + nt * 16 + m] =
            tanh_fast(accA[nt][r] + accB[nt][r]);

    // Reload h in A layout (transpose through LDS), convert to f16.
    // DS ops are in-order within a wave; no barrier needed (wave-private region).
#pragma unroll
    for (int kc = 0; kc < 2; ++kc)
      hA[kc] = load16_cvt_f16(hbuf + m * LDS_STRIDE + kc * 32 + kg * 16);

    // Rotate pipeline registers.
    xA = xA_next;
#pragma unroll
    for (int nt = 0; nt < 4; ++nt) xpart[nt] = xpart_next[nt];
  }

  // ---- head: out[b] = h_T[b] . W_fc + b_fc ----
  float partial = 0.0f;
#pragma unroll
  for (int j = 0; j < 32; ++j)
    partial += hbuf[m * LDS_STRIDE + kg * 32 + j] * W_fc[kg * 32 + j];
  partial += __shfl_down(partial, 16, 32);  // combine the two K halves
  if (lane < 16) out[row0 + lane] = partial + b_fc[0];
}

extern "C" void kernel_launch(void* const* d_in, const int* in_sizes, int n_in,
                              void* d_out, int out_size, void* d_ws, size_t ws_size,
                              hipStream_t stream) {
  const float* xp  = (const float*)d_in[0];
  const float* Wih = (const float*)d_in[1];
  const float* Whh = (const float*)d_in[2];
  const float* bih = (const float*)d_in[3];
  const float* bhh = (const float*)d_in[4];
  const float* Wfc = (const float*)d_in[5];
  const float* bfc = (const float*)d_in[6];
  float* outp = (float*)d_out;

  const int tiles  = RNN_BATCH / ROWS_PER_WAVE;     // 256 independent wave tasks
  const int blocks = tiles / WAVES_PER_BLOCK;       // 64 blocks x 128 threads
  rnn_fused_kernel<<<blocks, WAVES_PER_BLOCK * 32, 0, stream>>>(
      xp, Wih, Whh, bih, bhh, Wfc, bfc, outp);
}
